// MultiHeadAttention_9242769621942
// MI455X (gfx1250) — compile-verified, matrix operand order checked by reading, not run on hardware
//
#include <hip/hip_runtime.h>
#include <hip/hip_bf16.h>

#define DEVINL __device__ __forceinline__

typedef __attribute__((ext_vector_type(16))) __bf16 v16bf;
typedef __attribute__((ext_vector_type(8)))  __bf16 v8bf;
typedef __attribute__((ext_vector_type(8)))  float  v8f;

constexpr int B_ = 4, L_ = 2048, DM = 1024, NH = 16, DK = 64, DV = 64, DO = 1024;
constexpr int M_ROWS = B_ * L_; // 8192

// ---- WMMA helpers -----------------------------------------------------------
DEVINL v16bf frag_ld(const __bf16* p0) {
  // 16-bit A/B fragment: lane holds two 8-element (16B) chunks 16 elems apart
  v8bf lo = *(const v8bf*)p0;
  v8bf hi = *(const v8bf*)(p0 + 16);
  return __builtin_shufflevector(lo, hi, 0,1,2,3,4,5,6,7,8,9,10,11,12,13,14,15);
}
DEVINL v16bf frag_ld_g(const __bf16* base, int stride, int row, int c0) {
  return frag_ld(base + (size_t)row * stride + c0);
}
DEVINL v8f wmma_bf16(v16bf a, v16bf b, v8f c) {
  return __builtin_amdgcn_wmma_f32_16x16x32_bf16(false, a, false, b, (short)0, c,
                                                 false, false);
}

// ---- precision conversion kernels ------------------------------------------
__global__ __launch_bounds__(256) void cvt_q_kernel(const float* __restrict__ q,
                                                    __bf16* __restrict__ o, int n) {
  int i = blockIdx.x * 256 + threadIdx.x;
  if (i < n) o[i] = (__bf16)q[i];
}

// w_bf layout: [sel(q,k,v)][head][n=64][k=1024]  (B^T, contiguous K rows)
__global__ __launch_bounds__(256) void cvt_w_kernel(const float* __restrict__ wq,
                                                    const float* __restrict__ wk,
                                                    const float* __restrict__ wv,
                                                    __bf16* __restrict__ o) {
  int i = blockIdx.x * 256 + threadIdx.x;         // over 3*16*64*1024
  int k = i & 1023;
  int n = (i >> 10) & 63;
  int h = (i >> 16) & 15;
  int s = i >> 20;
  const float* src = (s == 0) ? wq : (s == 1) ? wk : wv;
  float v = src[((size_t)h * DM + k) * DK + n];
  if (s == 0) v *= 0.03125f;                      // fold 1/sqrt(d_model)=1/32
  o[i] = (__bf16)v;
}

__global__ __launch_bounds__(256) void cvt_pw_kernel(const float* __restrict__ pw,
                                                     __bf16* __restrict__ o, int n) {
  int i = blockIdx.x * 256 + threadIdx.x;
  if (i < n) o[i] = (__bf16)pw[i];               // proj_w is already [N][K]
}

// ---- QKV projection GEMM ----------------------------------------------------
// grid = (M/128, 3*NH); block = 256 (8 waves, 4 along M x 2 along N=64)
__global__ __launch_bounds__(256)
void qkv_gemm_kernel(const __bf16* __restrict__ qbf, const __bf16* __restrict__ wbf,
                     __bf16* __restrict__ qh, __bf16* __restrict__ kh,
                     __bf16* __restrict__ vt) {
  const int wave = threadIdx.x >> 5, lane = threadIdx.x & 31;
  const int lr = lane & 15, g = lane >> 4;
  const int m0 = blockIdx.x * 128 + (wave >> 1) * 32;
  const int n0 = (wave & 1) * 32;
  const int s = blockIdx.y / NH, h = blockIdx.y % NH;
  const __bf16* A  = qbf;
  const __bf16* Bt = wbf + (size_t)blockIdx.y * DK * DM;
  v8f acc[2][2] = {};
  #pragma unroll 4
  for (int kk = 0; kk < DM; kk += 32) {
    const int c = kk + g * 8;
    if (kk + 256 < DM) {
      __builtin_prefetch(A + (size_t)(m0 + lr) * DM + kk + 256, 0, 3);
    }
    v16bf a0 = frag_ld_g(A, DM, m0 + lr, c);
    v16bf a1 = frag_ld_g(A, DM, m0 + 16 + lr, c);
    v16bf b0 = frag_ld_g(Bt, DM, n0 + lr, c);
    v16bf b1 = frag_ld_g(Bt, DM, n0 + 16 + lr, c);
    acc[0][0] = wmma_bf16(a0, b0, acc[0][0]);
    acc[0][1] = wmma_bf16(a0, b1, acc[0][1]);
    acc[1][0] = wmma_bf16(a1, b0, acc[1][0]);
    acc[1][1] = wmma_bf16(a1, b1, acc[1][1]);
  }
  if (s == 2) {                                  // V stored transposed [h][b][dv][L]
    #pragma unroll
    for (int ti = 0; ti < 2; ++ti)
      #pragma unroll
      for (int tj = 0; tj < 2; ++tj)
        #pragma unroll
        for (int v = 0; v < 8; ++v) {
          int M = m0 + ti * 16 + v + 8 * g;
          int n = n0 + tj * 16 + lr;
          int b = M >> 11, l = M & (L_ - 1);
          vt[(((size_t)h * B_ + b) * DV + n) * L_ + l] = (__bf16)acc[ti][tj][v];
        }
  } else {                                       // Q/K stored [h][b][L][dk]
    __bf16* dst = (s == 0) ? qh : kh;
    #pragma unroll
    for (int ti = 0; ti < 2; ++ti)
      #pragma unroll
      for (int tj = 0; tj < 2; ++tj)
        #pragma unroll
        for (int v = 0; v < 8; ++v) {
          int M = m0 + ti * 16 + v + 8 * g;
          int n = n0 + tj * 16 + lr;
          int b = M >> 11, l = M & (L_ - 1);
          dst[(((size_t)h * B_ + b) * L_ + l) * DK + n] = (__bf16)acc[ti][tj][v];
        }
  }
}

// ---- flash attention --------------------------------------------------------
// grid = (L/128, NH*B); block = 256; wave w owns query rows [blk*128+w*16, +16)
__global__ __launch_bounds__(256)
void attn_kernel(const __bf16* __restrict__ qh, const __bf16* __restrict__ kh,
                 const __bf16* __restrict__ vt, __bf16* __restrict__ attn_out) {
  __shared__ __align__(16) __bf16 lds_p[128 * 64];   // per-wave 16x64 P tiles
  const int wave = threadIdx.x >> 5, lane = threadIdx.x & 31;
  const int lr = lane & 15, g = lane >> 4;
  const int h = blockIdx.y >> 2, b = blockIdx.y & 3;
  const size_t hb = (size_t)blockIdx.y;              // == h*4+b
  const __bf16* Q = qh + hb * (size_t)(L_ * DK);
  const __bf16* K = kh + hb * (size_t)(L_ * DK);
  const __bf16* V = vt + hb * (size_t)(DV * L_);
  const int qrow = blockIdx.x * 128 + wave * 16;

  v16bf qf0 = frag_ld_g(Q, DK, qrow + lr, 0 + g * 8);
  v16bf qf1 = frag_ld_g(Q, DK, qrow + lr, 32 + g * 8);

  float mst[8], lst[8];
  v8f oacc[4] = {};
  #pragma unroll
  for (int v = 0; v < 8; ++v) { mst[v] = -3.0e38f; lst[v] = 0.0f; }
  __bf16* pl = lds_p + wave * 16 * 64;

  for (int nb = 0; nb < L_; nb += 64) {
    // S = Q K^T (pre-scaled): 16 x 64 per wave
    v8f sacc[4];
    #pragma unroll
    for (int nt = 0; nt < 4; ++nt) {
      int key = nb + nt * 16 + lr;
      v16bf b0 = frag_ld_g(K, DK, key, 0 + g * 8);
      v16bf b1 = frag_ld_g(K, DK, key, 32 + g * 8);
      v8f t = {};
      t = wmma_bf16(qf0, b0, t);
      t = wmma_bf16(qf1, b1, t);
      sacc[nt] = t;
    }
    // online softmax: row stats live replicated across the 16 lanes of a half-wave
    float mx[8];
    #pragma unroll
    for (int v = 0; v < 8; ++v)
      mx[v] = fmaxf(fmaxf(sacc[0][v], sacc[1][v]), fmaxf(sacc[2][v], sacc[3][v]));
    #pragma unroll
    for (int off = 8; off > 0; off >>= 1)
      #pragma unroll
      for (int v = 0; v < 8; ++v)
        mx[v] = fmaxf(mx[v], __shfl_xor(mx[v], off, 32));
    #pragma unroll
    for (int v = 0; v < 8; ++v) {
      float mn = fmaxf(mst[v], mx[v]);
      float sc = __expf(mst[v] - mn);
      mst[v] = mn;
      lst[v] *= sc;
      #pragma unroll
      for (int jt = 0; jt < 4; ++jt) oacc[jt][v] *= sc;
    }
    float rs[8];
    #pragma unroll
    for (int v = 0; v < 8; ++v) rs[v] = 0.0f;
    #pragma unroll
    for (int nt = 0; nt < 4; ++nt)
      #pragma unroll
      for (int v = 0; v < 8; ++v) {
        float p = __expf(sacc[nt][v] - mst[v]);
        sacc[nt][v] = p;
        rs[v] += p;
      }
    #pragma unroll
    for (int off = 8; off > 0; off >>= 1)
      #pragma unroll
      for (int v = 0; v < 8; ++v)
        rs[v] += __shfl_xor(rs[v], off, 32);
    #pragma unroll
    for (int v = 0; v < 8; ++v) lst[v] += rs[v];

    // P (C-layout) -> LDS -> A-layout fragments (wave-private region)
    #pragma unroll
    for (int nt = 0; nt < 4; ++nt)
      #pragma unroll
      for (int v = 0; v < 8; ++v)
        pl[(v + 8 * g) * 64 + nt * 16 + lr] = (__bf16)sacc[nt][v];

    // O += P * V
    #pragma unroll
    for (int kt = 0; kt < 2; ++kt) {
      v16bf pf = frag_ld_g(pl, 64, lr, kt * 32 + g * 8);
      #pragma unroll
      for (int jt = 0; jt < 4; ++jt) {
        v16bf vf = frag_ld_g(V, L_, jt * 16 + lr, nb + kt * 32 + g * 8);
        oacc[jt] = wmma_bf16(pf, vf, oacc[jt]);
      }
    }
  }
  // normalize and write head-concat layout [b*L + l][h*64 + d]
  #pragma unroll
  for (int v = 0; v < 8; ++v) {
    float inv = 1.0f / lst[v];
    int row = b * L_ + qrow + v + 8 * g;
    #pragma unroll
    for (int jt = 0; jt < 4; ++jt) {
      int col = h * DV + jt * 16 + lr;
      attn_out[(size_t)row * (NH * DV) + col] = (__bf16)(oacc[jt][v] * inv);
    }
  }
}

// ---- output projection + bias + residual -----------------------------------
// grid = (M/128, DO/64); block = 256
__global__ __launch_bounds__(256)
void proj_gemm_kernel(const __bf16* __restrict__ attn, const __bf16* __restrict__ pw,
                      const float* __restrict__ pb, const float* __restrict__ q,
                      float* __restrict__ y) {
  const int wave = threadIdx.x >> 5, lane = threadIdx.x & 31;
  const int lr = lane & 15, g = lane >> 4;
  const int m0 = blockIdx.x * 128 + (wave >> 1) * 32;
  const int n0 = blockIdx.y * 64 + (wave & 1) * 32;
  v8f acc[2][2] = {};
  #pragma unroll 4
  for (int kk = 0; kk < DM; kk += 32) {
    const int c = kk + g * 8;
    if (kk + 256 < DM) {
      __builtin_prefetch(attn + (size_t)(m0 + lr) * DM + kk + 256, 0, 3);
    }
    v16bf a0 = frag_ld_g(attn, DM, m0 + lr, c);
    v16bf a1 = frag_ld_g(attn, DM, m0 + 16 + lr, c);
    v16bf b0 = frag_ld_g(pw, DM, n0 + lr, c);
    v16bf b1 = frag_ld_g(pw, DM, n0 + 16 + lr, c);
    acc[0][0] = wmma_bf16(a0, b0, acc[0][0]);
    acc[0][1] = wmma_bf16(a0, b1, acc[0][1]);
    acc[1][0] = wmma_bf16(a1, b0, acc[1][0]);
    acc[1][1] = wmma_bf16(a1, b1, acc[1][1]);
  }
  #pragma unroll
  for (int ti = 0; ti < 2; ++ti)
    #pragma unroll
    for (int tj = 0; tj < 2; ++tj)
      #pragma unroll
      for (int v = 0; v < 8; ++v) {
        int M = m0 + ti * 16 + v + 8 * g;
        int n = n0 + tj * 16 + lr;
        y[(size_t)M * DO + n] = acc[ti][tj][v] + pb[n] + q[(size_t)M * DM + n];
      }
}

// ---- LayerNorm (torch-style: unbiased std, divide by sigma+eps) -------------
__global__ __launch_bounds__(256)
void ln_kernel(const float* __restrict__ y, const float* __restrict__ ga,
               const float* __restrict__ gb, float* __restrict__ out) {
  const int row = blockIdx.x;
  const int t = threadIdx.x;
  const float* yr = y + (size_t)row * DO;
  float x[4], s = 0.0f, ss = 0.0f;
  #pragma unroll
  for (int i = 0; i < 4; ++i) {
    x[i] = yr[t + i * 256];
    s += x[i];
    ss += x[i] * x[i];
  }
  #pragma unroll
  for (int off = 16; off > 0; off >>= 1) {
    s  += __shfl_xor(s, off, 32);
    ss += __shfl_xor(ss, off, 32);
  }
  __shared__ float shs[8], shss[8];
  int wave = t >> 5, lane = t & 31;
  if (lane == 0) { shs[wave] = s; shss[wave] = ss; }
  __syncthreads();
  if (t == 0) {
    float ts = 0.0f, tss = 0.0f;
    #pragma unroll
    for (int i = 0; i < 8; ++i) { ts += shs[i]; tss += shss[i]; }
    shs[0] = ts; shss[0] = tss;
  }
  __syncthreads();
  float mu  = shs[0] * (1.0f / DO);
  float var = (shss[0] - (float)DO * mu * mu) * (1.0f / (DO - 1));
  float sig = sqrtf(fmaxf(var, 0.0f));
  float inv = 1.0f / (sig + 1e-3f);
  float* outr = out + (size_t)row * DO;
  #pragma unroll
  for (int i = 0; i < 4; ++i) {
    int n = t + i * 256;
    outr[n] = (x[i] - mu) * inv * ga[n] + gb[n];
  }
}

// ---- host-side launch -------------------------------------------------------
extern "C" void kernel_launch(void* const* d_in, const int* in_sizes, int n_in,
                              void* d_out, int out_size, void* d_ws, size_t ws_size,
                              hipStream_t stream) {
  const float* q  = (const float*)d_in[0];
  const float* wq = (const float*)d_in[1];
  const float* wk = (const float*)d_in[2];
  const float* wv = (const float*)d_in[3];
  const float* pw = (const float*)d_in[4];
  const float* pb = (const float*)d_in[5];
  const float* ga = (const float*)d_in[6];
  const float* gb = (const float*)d_in[7];
  float* out = (float*)d_out;
  (void)in_sizes; (void)n_in; (void)out_size; (void)ws_size;

  char* ws = (char*)d_ws;
  size_t off = 0;
  auto alloc = [&](size_t bytes) -> void* {
    void* p = ws + off;
    off += (bytes + 255) & ~(size_t)255;
    return p;
  };
  __bf16* qbf = (__bf16*)alloc((size_t)M_ROWS * DM * 2);          // 16 MB
  __bf16* wbf = (__bf16*)alloc((size_t)3 * NH * DK * DM * 2);     //  6 MB
  __bf16* qhh = (__bf16*)alloc((size_t)NH * B_ * L_ * DK * 2);    // 16 MB
  __bf16* khh = (__bf16*)alloc((size_t)NH * B_ * L_ * DK * 2);    // 16 MB
  __bf16* vtt = (__bf16*)alloc((size_t)NH * B_ * DV * L_ * 2);    // 16 MB
  __bf16* abf = (__bf16*)alloc((size_t)M_ROWS * (NH * DV) * 2);   // 16 MB
  __bf16* pwb = (__bf16*)alloc((size_t)DO * DM * 2);              //  2 MB
  float*  ybuf = (float*)alloc((size_t)M_ROWS * DO * 4);          // 32 MB

  cvt_q_kernel<<<(M_ROWS * DM) / 256, 256, 0, stream>>>(q, qbf, M_ROWS * DM);
  cvt_w_kernel<<<(3 * NH * DK * DM) / 256, 256, 0, stream>>>(wq, wk, wv, wbf);
  cvt_pw_kernel<<<(DO * DM) / 256, 256, 0, stream>>>(pw, pwb, DO * DM);

  dim3 g1(M_ROWS / 128, 3 * NH);
  qkv_gemm_kernel<<<g1, 256, 0, stream>>>(qbf, wbf, qhh, khh, vtt);

  dim3 g2(L_ / 128, NH * B_);
  attn_kernel<<<g2, 256, 0, stream>>>(qhh, khh, vtt, abf);

  dim3 g3(M_ROWS / 128, DO / 64);
  proj_gemm_kernel<<<g3, 256, 0, stream>>>(abf, pwb, pb, q, ybuf);

  ln_kernel<<<M_ROWS, 256, 0, stream>>>(ybuf, ga, gb, out);
}
